// VLRGAT_67903432950248
// MI455X (gfx1250) — matrix operand order repaired and checked by reading, not compiled, not run on hardware
//
#include <hip/hip_runtime.h>
#include <hip/hip_bf16.h>
#include <math.h>

typedef __attribute__((ext_vector_type(16))) _Float16 v16h;
typedef __attribute__((ext_vector_type(8)))  _Float16 v8h;
typedef __attribute__((ext_vector_type(4)))  _Float16 v4h;
typedef __attribute__((ext_vector_type(8)))  float    v8f;

#define CB_TOT 256   // C*B
#define NN     256   // nodes per graph
#define DD     128   // d_model
#define NH     4     // heads
#define DV     64    // d_kv
#define HD     256   // NH*DV

// Padded LDS row pitches (halves) to break bank conflicts; all give 16B-multiple pitch.
#define LDQ    72    // 64+8   (144 B pitch)
#define LDX    136   // 128+8  (272 B pitch)
#define LDW    264   // 256+8  (528 B pitch)
#define LDO    68    // f32 64+4 (272 B pitch)

// ---------- CDNA5 async global->LDS (ASYNCcnt path), guarded with fallback ----------
#if defined(__has_builtin)
#  if __has_builtin(__builtin_amdgcn_global_load_async_to_lds_b128)
#    define HAVE_ASYNC_LDS 1
#  endif
#endif

#ifdef HAVE_ASYNC_LDS
typedef int b128_t __attribute__((vector_size(4 * sizeof(int))));   // 16-byte payload type
typedef __attribute__((address_space(1))) b128_t* g_b128_p;          // global src
typedef __attribute__((address_space(3))) b128_t* l_b128_p;          // LDS dst
__device__ __forceinline__ void async_cp16(void* gsrc, void* ldst) {
  __builtin_amdgcn_global_load_async_to_lds_b128((g_b128_p)gsrc, (l_b128_p)ldst, 0, 0);
}
__device__ __forceinline__ void wait_async0() {
#  if __has_builtin(__builtin_amdgcn_s_wait_asynccnt)
  __builtin_amdgcn_s_wait_asynccnt(0);
#  else
  asm volatile("s_wait_asynccnt 0x0" ::: "memory");
#  endif
}
#endif

// ---------- WMMA fragment helpers (AMD table k order; layouts per ISA 7.12.2, wave32) ----------

// A fragment 16x32 f16 from row-major src: A[r][k] = src[(row0+r)*ld + k0+k]
__device__ __forceinline__ v16h ld_a(const _Float16* src, int ld, int row0, int k0, int lane) {
  const _Float16* p = src + (size_t)(row0 + (lane & 15)) * ld + k0 + ((lane & 16) ? 8 : 0);
  v8h lo = *(const v8h*)p;         // K kb..kb+7
  v8h hi = *(const v8h*)(p + 16);  // K kb+16..kb+23
  return __builtin_shufflevector(lo, hi, 0,1,2,3,4,5,6,7,8,9,10,11,12,13,14,15);
}

// B fragment 32x16 f16 where B = srcT: B[k][n] = src[(col0+n)*ld + k0+k]  (contiguous per lane)
__device__ __forceinline__ v16h ld_b_t(const _Float16* src, int ld, int k0, int col0, int lane) {
  const _Float16* p = src + (size_t)(col0 + (lane & 15)) * ld + k0 + ((lane & 16) ? 8 : 0);
  v8h lo = *(const v8h*)p;
  v8h hi = *(const v8h*)(p + 16);
  return __builtin_shufflevector(lo, hi, 0,1,2,3,4,5,6,7,8,9,10,11,12,13,14,15);
}

__device__ __forceinline__ v8f wmma_f16(v16h a, v16h b, v8f c) {
  return __builtin_amdgcn_wmma_f32_16x16x32_f16(false, a, false, b, (short)0, c, false, false);
}

// D store row-major (strided, epilogue-only)
__device__ __forceinline__ void st_d_f16(_Float16* dst, int ld, int row0, int col0, v8f d, int lane) {
  _Float16* p = dst + (size_t)(row0 + ((lane & 16) ? 8 : 0)) * ld + col0 + (lane & 15);
#pragma unroll
  for (int r = 0; r < 8; ++r) p[(size_t)r * ld] = (_Float16)d[r];
}
__device__ __forceinline__ void st_d_f32(float* dst, int ld, int row0, int col0, v8f d, int lane) {
  float* p = dst + (size_t)(row0 + ((lane & 16) ? 8 : 0)) * ld + col0 + (lane & 15);
#pragma unroll
  for (int r = 0; r < 8; ++r) p[(size_t)r * ld] = d[r];
}
// Transposed D store: dst[(col0+n)*ld + row0+r] -> 8 CONTIGUOUS halves per lane = one b128 store
__device__ __forceinline__ void st_d_f16_t(_Float16* dst, int ld, int row0, int col0, v8f d, int lane) {
  _Float16* p = dst + (size_t)(col0 + (lane & 15)) * ld + row0 + ((lane & 16) ? 8 : 0);
  v8h h;
#pragma unroll
  for (int r = 0; r < 8; ++r) h[r] = (_Float16)d[r];
  *(v8h*)p = h;
}

// ---------------- Kernel 1: fused QKV projection GEMM ----------------
// grid.x: 128-row M tile; grid.y in [0,12): wsel = y>>2 (Q/K/V), head = y&3.
// Q,K stored [cbh][n][dv] row-major; V stored TRANSPOSED [cbh][dv][n].
__global__ void qkv_gemm(const float* __restrict__ x,
                         const float* __restrict__ Wq,
                         const float* __restrict__ Wk,
                         const float* __restrict__ Wv,
                         _Float16* __restrict__ qws,
                         _Float16* __restrict__ kws,
                         _Float16* __restrict__ vtws) {
  extern __shared__ char smem[];
  _Float16* Xs = (_Float16*)smem;                        // [128][LDX]
  _Float16* Wt = (_Float16*)(smem + 128 * LDX * 2);      // [64][LDX]  (W^T tile)

  const int t = threadIdx.x, lane = t & 31, wv = t >> 5;
  const int m0   = blockIdx.x * 128;
  const int h    = blockIdx.y & 3;
  const int wsel = blockIdx.y >> 2;
  const float* W = (wsel == 0) ? Wq : (wsel == 1 ? Wk : Wv);
  const int colbase = h * DV;
  __builtin_prefetch(W + colbase, 0, 3);                 // global_prefetch_b8 (L2-resident weights)

  for (int i = t * 4; i < 128 * 128; i += 1024) {        // X tile f32->f16 (coalesced)
    const int r = i >> 7, c = i & 127;
    const float4 v = *(const float4*)(x + (size_t)(m0 + r) * DD + c);
    v4h o; o[0]=(_Float16)v.x; o[1]=(_Float16)v.y; o[2]=(_Float16)v.z; o[3]=(_Float16)v.w;
    *(v4h*)(Xs + (size_t)r * LDX + c) = o;
  }
  for (int i = t; i < 64 * 128; i += 256) {              // W^T tile: coalesced reads, padded writes
    const int j = i & 63, k = i >> 6;                    // j: out col, k: reduction
    Wt[(size_t)j * LDX + k] = (_Float16)W[(size_t)k * HD + colbase + j];
  }
  __syncthreads();

  const int mw = (wv >> 1) * 32;   // 4 waves down M
  const int nw = (wv & 1) * 32;    // 2 waves across N
  v8f acc[2][2] = {};
  for (int k0 = 0; k0 < 128; k0 += 32) {
    v16h a0 = ld_a(Xs, LDX, mw,      k0, lane);
    v16h a1 = ld_a(Xs, LDX, mw + 16, k0, lane);
    v16h b0 = ld_b_t(Wt, LDX, k0, nw,      lane);
    v16h b1 = ld_b_t(Wt, LDX, k0, nw + 16, lane);
    acc[0][0] = wmma_f16(a0, b0, acc[0][0]);
    acc[0][1] = wmma_f16(a0, b1, acc[0][1]);
    acc[1][0] = wmma_f16(a1, b0, acc[1][0]);
    acc[1][1] = wmma_f16(a1, b1, acc[1][1]);
  }

  const int cb = m0 >> 8;
  const int n0 = m0 & 255;
  const size_t cbh = (size_t)(cb * NH + h);
  if (wsel < 2) {
    _Float16* out = (wsel == 0 ? qws : kws) + (cbh * NN + n0) * DV;
#pragma unroll
    for (int ti = 0; ti < 2; ++ti)
#pragma unroll
      for (int tj = 0; tj < 2; ++tj)
        st_d_f16(out, DV, mw + ti * 16, nw + tj * 16, acc[ti][tj], lane);
  } else {
    // V transposed: vt[cbh][d][n]; transposed store = contiguous b128 global stores
    _Float16* out = vtws + cbh * DV * NN + n0;           // col index = n (local row)
#pragma unroll
    for (int ti = 0; ti < 2; ++ti)
#pragma unroll
      for (int tj = 0; tj < 2; ++tj)
        st_d_f16_t(out, NN, mw + ti * 16, nw + tj * 16, acc[ti][tj], lane);
  }
}

// ---------------- Kernel 2: attention + V^T P V per (cb, h) ----------------
// LDS: Qs[256][LDQ] (reused as Vt[64][LDW]), Ks[256][LDQ] (reused as O1[64][LDW]), St[256][LDW].
__global__ void attn_kernel(_Float16* __restrict__ qws,
                            _Float16* __restrict__ kws,
                            _Float16* __restrict__ vtws,
                            _Float16* __restrict__ aws) {
  extern __shared__ char smem[];
  _Float16* Qs = (_Float16*)smem;                        // 36864 B
  _Float16* Ks = (_Float16*)(smem + 36864);              // 36864 B
  _Float16* St = (_Float16*)(smem + 73728);              // 256*LDW*2 = 135168 B

  const int t = threadIdx.x, lane = t & 31, wv = t >> 5;
  const int cbh = blockIdx.x;
  const size_t base = (size_t)cbh * NN * DV;

  {                                                      // stage Q, K into padded rows
#ifdef HAVE_ASYNC_LDS
    v8h* qsrc = (v8h*)(qws + base);
    v8h* ksrc = (v8h*)(kws + base);
    for (int i = t; i < 2048; i += 256) {
      const int n = i >> 3, c = i & 7;                   // 8 x v8h per 64-half row
      async_cp16(qsrc + i, Qs + (size_t)n * LDQ + c * 8);
      async_cp16(ksrc + i, Ks + (size_t)n * LDQ + c * 8);
    }
    wait_async0();
#else
    const v8h* qsrc = (const v8h*)(qws + base);
    const v8h* ksrc = (const v8h*)(kws + base);
    for (int i = t; i < 2048; i += 256) {
      const int n = i >> 3, c = i & 7;
      *(v8h*)(Qs + (size_t)n * LDQ + c * 8) = qsrc[i];
      *(v8h*)(Ks + (size_t)n * LDQ + c * 8) = ksrc[i];
    }
#endif
  }
  __syncthreads();

  // St = (Q K^T)^T * (1/sqrt(dv)) : transposed store -> contiguous ds_store_b128
  for (int rt = wv * 32; rt < wv * 32 + 32; rt += 16) {
    v16h a0 = ld_a(Qs, LDQ, rt, 0,  lane);
    v16h a1 = ld_a(Qs, LDQ, rt, 32, lane);
    for (int ct = 0; ct < NN; ct += 16) {
      v16h b0 = ld_b_t(Ks, LDQ, 0,  ct, lane);
      v16h b1 = ld_b_t(Ks, LDQ, 32, ct, lane);
      v8f s = {};
      s = wmma_f16(a0, b0, s);
      s = wmma_f16(a1, b1, s);
#pragma unroll
      for (int r = 0; r < 8; ++r) s[r] *= 0.125f;
      st_d_f16_t(St, LDW, rt, ct, s, lane);              // St[c][r] = S[r][c]
    }
  }
  __syncthreads();

  {                                                      // Vt DMA into Qs region (Q dead),
                                                         // overlapped with softmax below
    _Float16* Vt = Qs;                                   // [64][LDW]
#ifdef HAVE_ASYNC_LDS
    v8h* vsrc = (v8h*)(vtws + base);                     // [64][256] contiguous
    for (int i = t; i < 2048; i += 256) {
      const int d = i >> 5, c = i & 31;
      async_cp16(vsrc + i, Vt + (size_t)d * LDW + c * 8);
    }
#else
    const v8h* vsrc = (const v8h*)(vtws + base);
    for (int i = t; i < 2048; i += 256) {
      const int d = i >> 5, c = i & 31;
      *(v8h*)(Vt + (size_t)d * LDW + c * 8) = vsrc[i];
    }
#endif
  }
  {                                                      // softmax: thread t = query row t (col t of St)
    _Float16* col = St + t;
    float mx = -3.0e38f;
    for (int j = 0; j < NN; ++j) mx = fmaxf(mx, (float)col[(size_t)j * LDW]);
    float sum = 0.f;
    for (int j = 0; j < NN; ++j) sum += __expf((float)col[(size_t)j * LDW] - mx);
    const float inv = 1.f / sum;
    for (int j = 0; j < NN; ++j)
      col[(size_t)j * LDW] = (_Float16)(__expf((float)col[(size_t)j * LDW] - mx) * inv);
  }
#ifdef HAVE_ASYNC_LDS
  wait_async0();                                         // Vt DMA done (overlapped with softmax)
#endif
  __syncthreads();

  _Float16* Vt = Qs;                                     // [64][LDW]
  _Float16* O1 = Ks;                                     // [64][LDW]

  // O1 = V^T @ P : A = Vt rows (contig), B[n][m] = P[n][m] = St[m][n] (contig ld_b_t)
  for (int dt = 0; dt < DV; dt += 16) {
#pragma unroll
    for (int cc = 0; cc < 2; ++cc) {
      const int ct = (wv * 2 + cc) * 16;
      v8f o = {};
      for (int n0 = 0; n0 < NN; n0 += 32) {
        v16h a = ld_a(Vt, LDW, dt, n0, lane);
        v16h b = ld_b_t(St, LDW, n0, ct, lane);
        o = wmma_f16(a, b, o);
      }
      st_d_f16(O1, LDW, dt, ct, o, lane);
    }
  }
  __syncthreads();

  // O2 = O1 @ V : A = O1 rows (contig), B[m][e] = V[m][e] = Vt[e][m] (contig ld_b_t)
  const int cb = cbh >> 2, h = cbh & 3;
  _Float16* outp = aws + (size_t)cb * DV * HD + h * DV;  // [cb][64][256] staging
  {
    const int dt = (wv >> 1) * 16;
#pragma unroll
    for (int cc = 0; cc < 2; ++cc) {
      const int et = ((wv & 1) * 2 + cc) * 16;
      v8f o = {};
      for (int m0 = 0; m0 < NN; m0 += 32) {
        v16h a = ld_a(O1, LDW, dt, m0, lane);
        v16h b = ld_b_t(Vt, LDW, m0, et, lane);
        o = wmma_f16(a, b, o);
      }
      st_d_f16(outp, HD, dt, et, o, lane);
    }
  }
}

// ---------------- Kernel 3: out = LN2(LN1([64,256] @ Wfc)) per cb ----------------
__global__ void fc_ln(_Float16* __restrict__ aws,
                      const float* __restrict__ Wfc,
                      const float* __restrict__ g1, const float* __restrict__ b1,
                      const float* __restrict__ g2, const float* __restrict__ b2,
                      float* __restrict__ out) {
  extern __shared__ char smem[];
  _Float16* Ys = (_Float16*)smem;                        // [64][LDW]  33792 B
  _Float16* Wt = (_Float16*)(smem + 64 * LDW * 2);       // [64][LDW]  (Wfc^T) 33792 B
  float*    Os = (float*)(smem + 2 * 64 * LDW * 2);      // [64][LDO]  17408 B

  const int t = threadIdx.x, lane = t & 31, wv = t >> 5;
  const int cb = blockIdx.x;
  __builtin_prefetch(Wfc, 0, 3);

  {                                                      // Ys DMA, overlapped with Wfc^T convert
#ifdef HAVE_ASYNC_LDS
    v8h* src = (v8h*)(aws + (size_t)cb * DV * HD);
    for (int i = t; i < 2048; i += 256) {
      const int r = i >> 5, c = i & 31;
      async_cp16(src + i, Ys + (size_t)r * LDW + c * 8);
    }
#else
    const v8h* src = (const v8h*)(aws + (size_t)cb * DV * HD);
    for (int i = t; i < 2048; i += 256) {
      const int r = i >> 5, c = i & 31;
      *(v8h*)(Ys + (size_t)r * LDW + c * 8) = src[i];
    }
#endif
  }
  for (int i = t; i < HD * DV; i += 256) {               // Wfc^T: coalesced reads, padded writes
    const int j = i & 63, k = i >> 6;
    Wt[(size_t)j * LDW + k] = (_Float16)Wfc[(size_t)k * DV + j];
  }
#ifdef HAVE_ASYNC_LDS
  wait_async0();
#endif
  __syncthreads();

  {
    const int dt = (wv >> 1) * 16;
#pragma unroll
    for (int cc = 0; cc < 2; ++cc) {
      const int et = ((wv & 1) * 2 + cc) * 16;
      v8f o = {};
      for (int k0 = 0; k0 < HD; k0 += 32) {
        v16h a = ld_a(Ys, LDW, dt, k0, lane);
        v16h b = ld_b_t(Wt, LDW, k0, et, lane);
        o = wmma_f16(a, b, o);
      }
      st_d_f32(Os, LDO, dt, et, o, lane);
    }
  }
  __syncthreads();

  if (t < DV) {                                          // fused LN1 + LN2 over last dim (64)
    const float* row = Os + (size_t)t * LDO;
    float mu = 0.f;
    for (int e = 0; e < DV; ++e) mu += row[e];
    mu *= (1.f / 64.f);
    float var = 0.f;
    for (int e = 0; e < DV; ++e) { const float d = row[e] - mu; var += d * d; }
    var *= (1.f / 64.f);
    const float rs = rsqrtf(var + 1e-6f);
    float tv[64];
    float mu2 = 0.f;
    for (int e = 0; e < DV; ++e) { tv[e] = (row[e] - mu) * rs * g1[e] + b1[e]; mu2 += tv[e]; }
    mu2 *= (1.f / 64.f);
    float var2 = 0.f;
    for (int e = 0; e < DV; ++e) { const float d = tv[e] - mu2; var2 += d * d; }
    var2 *= (1.f / 64.f);
    const float rs2 = rsqrtf(var2 + 1e-6f);
    const int c = cb >> 5, b = cb & 31;                  // cb = c*32 + b
    float* op = out + ((size_t)(b * 8 + c) * DV + t) * DV;
    for (int e = 0; e < DV; ++e) op[e] = (tv[e] - mu2) * rs2 * g2[e] + b2[e];
  }
}

// ---------------- host launch ----------------
extern "C" void kernel_launch(void* const* d_in, const int* in_sizes, int n_in,
                              void* d_out, int out_size, void* d_ws, size_t ws_size,
                              hipStream_t stream) {
  const float* x   = (const float*)d_in[0];
  const float* Wq  = (const float*)d_in[1];
  const float* Wk  = (const float*)d_in[2];
  const float* Wv  = (const float*)d_in[3];
  const float* Wfc = (const float*)d_in[4];
  const float* g1  = (const float*)d_in[5];
  const float* b1  = (const float*)d_in[6];
  const float* g2  = (const float*)d_in[7];
  const float* b2  = (const float*)d_in[8];
  float* out = (float*)d_out;

  const size_t qkv_elems = (size_t)CB_TOT * NH * NN * DV;  // 16.7M halves each
  _Float16* qws  = (_Float16*)d_ws;
  _Float16* kws  = qws + qkv_elems;
  _Float16* vtws = kws + qkv_elems;                        // V stored transposed per (cb,h)
  _Float16* aws  = vtws + qkv_elems;                       // [CB][64][256] f16

  const int smem1 = (128 * LDX + 64 * LDX) * 2;            // 52224
  const int smem2 = 2 * 256 * LDQ * 2 + 256 * LDW * 2;     // 208896
  const int smem3 = 2 * 64 * LDW * 2 + 64 * LDO * 4;       // 84992

  (void)hipFuncSetAttribute((const void*)attn_kernel, hipFuncAttributeMaxDynamicSharedMemorySize, smem2);
  (void)hipFuncSetAttribute((const void*)fc_ln,       hipFuncAttributeMaxDynamicSharedMemorySize, smem3);

  qkv_gemm   <<<dim3(512, 12), 256, smem1, stream>>>(x, Wq, Wk, Wv, qws, kws, vtws);
  attn_kernel<<<dim3(1024),    256, smem2, stream>>>(qws, kws, vtws, aws);
  fc_ln      <<<dim3(256),     256, smem3, stream>>>(aws, Wfc, g1, b1, g2, b2, out);
}
